// _WindowAttention3D_57535381897310
// MI455X (gfx1250) — matrix operand order repaired and checked by reading, not compiled, not run on hardware
//
#include <hip/hip_runtime.h>

typedef __attribute__((ext_vector_type(16))) _Float16 v16h;
typedef __attribute__((ext_vector_type(8)))  _Float16 v8h;
typedef __attribute__((ext_vector_type(4)))  _Float16 v4h;
typedef __attribute__((ext_vector_type(8)))  float    v8f;

#define C_DIM    192
#define QKV_ROWS 576
#define NQKVW   (QKV_ROWS * C_DIM)   // 110592
#define NPROJW  (C_DIM * C_DIM)      // 36864

// LDS layout (bytes)
#define H_OFF     0          // h / attn activation: 64*192 f16 = 24576 B
#define QKV_OFF   24576      // per head: q[64][32], k[64][32], vT[32][64] f16 -> 73728 B
#define S_OFF     98304      // scores: 6*64*64 f32 = 98304 B (reused as out f32 later)
#define P_OFF     196608     // probs:  6*64*64 f16 = 49152 B
#define RS_OFF    245760     // recip row sums: 6*64 f32 = 1536 B
#define LDS_BYTES 247296

// ---- WMMA fragment loaders -------------------------------------------------
// A matrix 16x32 f16: lane<16 holds K=c0..c0+15,
// lane>=16 holds K=c0+16..c0+31 (contiguous per lane-half); row m = lane&15 (+m0).
__device__ __forceinline__ v16h frag_a_lds(const _Float16* base, int m0, int stride,
                                           int c0, unsigned lane) {
  const _Float16* p = base + (m0 + (lane & 15)) * stride + c0 + ((lane >> 4) << 4);
  v8h lo = *(const v8h*)p;
  v8h hi = *(const v8h*)(p + 8);
  return __builtin_shufflevector(lo, hi, 0,1,2,3,4,5,6,7,8,9,10,11,12,13,14,15);
}
// B matrix 32x16 f16: lane<16 holds K=c0..c0+15 of column n, lane>=16 holds
// K=c0+16..c0+31 (contiguous per lane-half).
__device__ __forceinline__ v16h frag_b_lds(const _Float16* base, int n0, int stride,
                                           int c0, unsigned lane) {
  const _Float16* p = base + (n0 + (lane & 15)) * stride + c0 + ((lane >> 4) << 4);
  v8h lo = *(const v8h*)p;
  v8h hi = *(const v8h*)(p + 8);
  return __builtin_shufflevector(lo, hi, 0,1,2,3,4,5,6,7,8,9,10,11,12,13,14,15);
}
__device__ __forceinline__ v16h frag_b_glb(const _Float16* __restrict__ base, int n0,
                                           int stride, int c0, unsigned lane) {
  const _Float16* p = base + (n0 + (lane & 15)) * stride + c0 + ((lane >> 4) << 4);
  v8h lo = *(const v8h*)p;
  v8h hi = *(const v8h*)(p + 8);
  return __builtin_shufflevector(lo, hi, 0,1,2,3,4,5,6,7,8,9,10,11,12,13,14,15);
}

#define WMMA(A, B, Cacc) \
  __builtin_amdgcn_wmma_f32_16x16x32_f16(false, (A), false, (B), (short)0, (Cacc), false, false)

// ---- weight conversion: f32 -> f16 -----------------------------------------
__global__ void _wa3d_convert(const float* __restrict__ qkv_w,
                              const float* __restrict__ proj_w,
                              _Float16* __restrict__ wq,
                              _Float16* __restrict__ wp) {
  int i = blockIdx.x * blockDim.x + threadIdx.x;
  if (i < NQKVW)  wq[i] = (_Float16)qkv_w[i];
  if (i < NPROJW) wp[i] = (_Float16)proj_w[i];
}

// ---- fused window attention: one block per 4x4x4 window --------------------
__global__ __launch_bounds__(256, 1)
void _wa3d_kernel(const float* __restrict__ x,
                  const float* __restrict__ qkv_b,
                  const float* __restrict__ proj_b,
                  const _Float16* __restrict__ wq,
                  const _Float16* __restrict__ wp,
                  float* __restrict__ out) {
  extern __shared__ char smem[];
  _Float16* hbuf = (_Float16*)(smem + H_OFF);     // [64][192]
  _Float16* qkvb = (_Float16*)(smem + QKV_OFF);   // per head: q|k|vT
  float*    Sbuf = (float*)(smem + S_OFF);        // [6][64][64]
  _Float16* Pbuf = (_Float16*)(smem + P_OFF);     // [6][64][64]
  float*    rsum = (float*)(smem + RS_OFF);       // [6][64]
  float*    obuf = (float*)(smem + S_OFF);        // reuse S: [64][192] f32

  const int tid  = threadIdx.x;
  const unsigned lane = tid & 31u;
  // wave index as a true scalar: makes all tile-derived values SGPRs,
  // turning the q/k/v store selection into scalar branches (no exec masking)
  const int wave = __builtin_amdgcn_readfirstlane(tid >> 5);   // 8 waves

  const int b    = blockIdx.x;
  const int nwx = b >> 8, nwy = (b >> 4) & 15, nwz = b & 15;
  const int x0 = nwx * 4, y0 = nwy * 4;
  const int z0 = nwz * 4 + 2;                     // shift folded in
  const bool zwrap = (z0 + 3) >= 64;              // only last z-window

  // ---- Phase A: gather shifted window -> hbuf (f16) ----
  {
    const int wxy = tid & 15;
    const int wx = wxy >> 2, wy = wxy & 3;
    const int gx = (x0 + wx + 2) & 63;
    const int gy = (y0 + wy + 2) & 63;
    for (int it = 0; it < 12; ++it) {
      int c = (tid >> 4) + 16 * it;               // 16 c-slots * 12 iters = 192
      const float* xp = x + (size_t)c * 262144 + gx * 4096 + gy * 64;
      float a0, a1, a2, a3;
      if (!zwrap) {
        float2 p0 = *(const float2*)(xp + z0);
        float2 p1 = *(const float2*)(xp + z0 + 2);
        a0 = p0.x; a1 = p0.y; a2 = p1.x; a3 = p1.y;
      } else {
        a0 = xp[z0 & 63]; a1 = xp[(z0 + 1) & 63];
        a2 = xp[(z0 + 2) & 63]; a3 = xp[(z0 + 3) & 63];
      }
      int tb = (wx * 16 + wy * 4) * 192 + c;
      hbuf[tb]       = (_Float16)a0;
      hbuf[tb + 192] = (_Float16)a1;
      hbuf[tb + 384] = (_Float16)a2;
      hbuf[tb + 576] = (_Float16)a3;
    }
  }
  __syncthreads();

  // ---- Phase B: QKV GEMM (64x192 @ 192x576), bias, scatter to q/k/vT ----
  const float scale = 0.17677669529663687f;       // 32^-0.5
#pragma unroll 1
  for (int i = 0; i < 18; ++i) {                  // 144 tiles / 8 waves
    int tile = wave + i * 8;                      // scalar
    int mt = tile & 3, nt = tile >> 2;
    int t0 = mt * 16, j0 = nt * 16;
    v8f acc = {};
#pragma unroll
    for (int kk = 0; kk < 6; ++kk) {
      v16h a  = frag_a_lds(hbuf, t0, 192, kk * 32, lane);
      v16h bf = frag_b_glb(wq, j0, 192, kk * 32, lane);
      acc = WMMA(a, bf, acc);
    }
    // which / head / d0 are scalar (tiles never straddle 32-boundaries)
    int which = j0 / 192;                         // 0=q, 1=k, 2=v
    int rem0  = j0 - which * 192;
    int head  = rem0 >> 5;
    int d     = (rem0 & 31) + (int)(lane & 15);
    float bias = qkv_b[j0 + (int)(lane & 15)];
    _Float16* qh = qkvb + head * 6144;
    int mbase = t0 + (int)((lane >> 4) << 3);
    if (which == 0) {
#pragma unroll
      for (int v = 0; v < 8; ++v)
        qh[(mbase + v) * 32 + d] = (_Float16)((acc[v] + bias) * scale);
    } else if (which == 1) {
#pragma unroll
      for (int v = 0; v < 8; ++v)
        qh[2048 + (mbase + v) * 32 + d] = (_Float16)(acc[v] + bias);
    } else {                                      // v stored transposed: vT[d][t]
#pragma unroll
      for (int v = 0; v < 8; ++v)
        qh[4096 + d * 64 + (mbase + v)] = (_Float16)(acc[v] + bias);
    }
  }
  __syncthreads();

  // ---- Phase C: S = q @ k^T, 6 heads * 16 tiles over 8 waves ----
#pragma unroll 1
  for (int i = 0; i < 12; ++i) {                  // 96 tiles / 8 waves
    int tile = wave + i * 8;                      // scalar
    int head = tile >> 4, sub = tile & 15;
    int mi = sub >> 2, ni = sub & 3;
    const _Float16* qh = qkvb + head * 6144;
    const _Float16* kh = qh + 2048;
    float* Sh = Sbuf + head * 4096;
    v16h a  = frag_a_lds(qh, mi * 16, 32, 0, lane);
    v16h bf = frag_b_lds(kh, ni * 16, 32, 0, lane);
    v8f acc = {};
    acc = WMMA(a, bf, acc);
    int mbase = mi * 16 + (int)((lane >> 4) << 3);
    int nn = ni * 16 + (int)(lane & 15);
#pragma unroll
    for (int v = 0; v < 8; ++v) Sh[(mbase + v) * 64 + nn] = acc[v];
  }
  __syncthreads();

  // ---- Phase D: row softmax -> P (f16, unnormalized) + reciprocal sums ----
  for (int idx = tid; idx < 384; idx += 256) {    // 6 heads * 64 rows
    int head = idx >> 6, r = idx & 63;
    const float4* row = (const float4*)(Sbuf + head * 4096 + r * 64);
    _Float16* Ph = Pbuf + head * 4096 + r * 64;
    float mx = -3.4e38f;
#pragma unroll
    for (int i = 0; i < 16; ++i) {
      float4 c4 = row[i];
      mx = fmaxf(mx, fmaxf(fmaxf(c4.x, c4.y), fmaxf(c4.z, c4.w)));
    }
    float sum = 0.f;
#pragma unroll
    for (int i = 0; i < 16; ++i) {
      float4 c4 = row[i];
      float e0 = __expf(c4.x - mx), e1 = __expf(c4.y - mx);
      float e2 = __expf(c4.z - mx), e3 = __expf(c4.w - mx);
      sum += (e0 + e1) + (e2 + e3);
      v4h pk = {(_Float16)e0, (_Float16)e1, (_Float16)e2, (_Float16)e3};
      *(v4h*)(Ph + i * 4) = pk;
    }
    rsum[head * 64 + r] = 1.0f / sum;
  }
  __syncthreads();

  // ---- Phase E: O = P @ V, scaled by 1/rowsum; 6 heads * 8 tiles / 8 waves --
#pragma unroll 1
  for (int i = 0; i < 6; ++i) {                   // 48 tiles / 8 waves
    int tile = wave + i * 8;                      // scalar
    int head = tile >> 3, sub = tile & 7;
    int mi = sub >> 1, di = sub & 1;
    const _Float16* vT = qkvb + head * 6144 + 4096;
    const _Float16* Ph = Pbuf + head * 4096;
    const float* rs = rsum + head * 64;
    v8f acc = {};
#pragma unroll
    for (int k2 = 0; k2 < 2; ++k2) {
      v16h a  = frag_a_lds(Ph, mi * 16, 64, k2 * 32, lane);
      v16h bf = frag_b_lds(vT, di * 16, 64, k2 * 32, lane);
      acc = WMMA(a, bf, acc);
    }
    int mbase = mi * 16 + (int)((lane >> 4) << 3);
    int cc = head * 32 + di * 16 + (int)(lane & 15);
#pragma unroll
    for (int v = 0; v < 8; ++v) {
      int m = mbase + v;
      hbuf[m * 192 + cc] = (_Float16)(acc[v] * rs[m]);
    }
  }
  __syncthreads();

  // ---- Phase F: proj GEMM (64x192 @ 192x192) + bias -> obuf (f32) ----
#pragma unroll 1
  for (int i = 0; i < 6; ++i) {                   // 48 tiles / 8 waves
    int tile = wave + i * 8;                      // scalar
    int mt = tile & 3, nt = tile >> 2;
    int t0 = mt * 16, j0 = nt * 16;
    v8f acc = {};
#pragma unroll
    for (int kk = 0; kk < 6; ++kk) {
      v16h a  = frag_a_lds(hbuf, t0, 192, kk * 32, lane);
      v16h bf = frag_b_glb(wp, j0, 192, kk * 32, lane);
      acc = WMMA(a, bf, acc);
    }
    int n = j0 + (int)(lane & 15);
    float bias = proj_b[n];
    int mbase = t0 + (int)((lane >> 4) << 3);
#pragma unroll
    for (int v = 0; v < 8; ++v) obuf[(mbase + v) * 192 + n] = acc[v] + bias;
  }
  __syncthreads();

  // ---- Phase G: scatter with inverse shift -> out ----
  {
    const int wxy = tid & 15;
    const int wx = wxy >> 2, wy = wxy & 3;
    const int gx = (x0 + wx + 2) & 63;
    const int gy = (y0 + wy + 2) & 63;
    for (int it = 0; it < 12; ++it) {
      int c = (tid >> 4) + 16 * it;
      float* op = out + (size_t)c * 262144 + gx * 4096 + gy * 64;
      int tb = (wx * 16 + wy * 4) * 192 + c;
      float a0 = obuf[tb], a1 = obuf[tb + 192];
      float a2 = obuf[tb + 384], a3 = obuf[tb + 576];
      if (!zwrap) {
        float2 p0; p0.x = a0; p0.y = a1;
        float2 p1; p1.x = a2; p1.y = a3;
        *(float2*)(op + z0) = p0;
        *(float2*)(op + z0 + 2) = p1;
      } else {
        op[z0 & 63] = a0; op[(z0 + 1) & 63] = a1;
        op[(z0 + 2) & 63] = a2; op[(z0 + 3) & 63] = a3;
      }
    }
  }
}

extern "C" void kernel_launch(void* const* d_in, const int* in_sizes, int n_in,
                              void* d_out, int out_size, void* d_ws, size_t ws_size,
                              hipStream_t stream) {
  const float* x      = (const float*)d_in[0];
  const float* qkv_w  = (const float*)d_in[1];
  const float* qkv_b  = (const float*)d_in[2];
  const float* proj_w = (const float*)d_in[3];
  const float* proj_b = (const float*)d_in[4];
  float* out = (float*)d_out;

  _Float16* wq = (_Float16*)d_ws;          // 576x192 f16
  _Float16* wp = wq + NQKVW;               // 192x192 f16

  _wa3d_convert<<<(NQKVW + 255) / 256, 256, 0, stream>>>(qkv_w, proj_w, wq, wp);

  (void)hipFuncSetAttribute((const void*)_wa3d_kernel,
                            hipFuncAttributeMaxDynamicSharedMemorySize, LDS_BYTES);
  _wa3d_kernel<<<4096, 256, LDS_BYTES, stream>>>(x, qkv_b, proj_b, wq, wp, out);
}
